// AttentionLayer_53317724013262
// MI455X (gfx1250) — matrix operand order repaired and checked by reading, not compiled, not run on hardware
//
#include <hip/hip_runtime.h>

// ---------------------------------------------------------------------------
// Problem constants (from reference)
// ---------------------------------------------------------------------------
static constexpr int Bn   = 2;
static constexpr int Sn   = 2048;
static constexpr int HIDn = 2048;
static constexpr int Hn   = 16;
static constexpr int KVHn = 4;
static constexpr int Dn   = 128;
static constexpr int Mrows = Bn * Sn;      // 4096
static constexpr int HD    = Hn * Dn;      // 2048
static constexpr int KVD   = KVHn * Dn;    // 512

typedef __attribute__((ext_vector_type(16))) _Float16 v16h;
typedef __attribute__((ext_vector_type(8)))  _Float16 v8h;
typedef __attribute__((ext_vector_type(8)))  float    v8f;

__device__ __forceinline__ v8f wmma16(v16h a, v16h b, v8f c) {
  // D(16x16,f32) = A(16x32,f16) x B(32x16,f16) + C
  return __builtin_amdgcn_wmma_f32_16x16x32_f16(false, a, false, b,
                                                (short)0, c, false, false);
}

// A-fragment (16x32 f16): lane holds row (lane&15); K chunks at
// 8*(lane>>4)+{0..7} and 16+8*(lane>>4)+{0..7}.  tile is row-major, ld halves.
__device__ __forceinline__ v16h load_fragA(const _Float16* tile, int ld, int lane) {
  int r  = lane & 15;
  int hb = (lane >> 4) * 8;
  const _Float16* p = tile + r * ld + hb;
  v8h lo = *(const v8h*)p;
  v8h hi = *(const v8h*)(p + 16);
  return __builtin_shufflevector(lo, hi, 0,1,2,3,4,5,6,7,8,9,10,11,12,13,14,15);
}

// B-fragment (32x16 f16): lane holds column n = lane&15; K chunks at
// 8*(lane>>4)+{0..7} and 16+8*(lane>>4)+{0..7}.
// Source stored "transposed": 16 rows indexed by n, K contiguous, ld halves.
__device__ __forceinline__ v16h load_fragB(const _Float16* tile, int ld, int lane) {
  int n  = lane & 15;
  int hb = (lane >> 4) * 8;
  const _Float16* p = tile + n * ld + hb;
  v8h lo = *(const v8h*)p;
  v8h hi = *(const v8h*)(p + 16);
  return __builtin_shufflevector(lo, hi, 0,1,2,3,4,5,6,7,8,9,10,11,12,13,14,15);
}

// CDNA5 async LDS-DMA: per-lane 16B global -> LDS, tracked by ASYNCcnt.
// lds_off = low 32 bits of the flat shared-aperture address (== LDS offset).
__device__ __forceinline__ void async_g2l_b128(const _Float16* lds_ptr,
                                               const _Float16* gptr) {
  unsigned lds_off = (unsigned)(size_t)lds_ptr;
  asm volatile("global_load_async_to_lds_b128 %0, %1, off"
               :: "v"(lds_off), "v"(gptr) : "memory");
}

// ---------------------------------------------------------------------------
// Elementwise convert / pack kernels
// ---------------------------------------------------------------------------
__global__ void conv_h(const float* __restrict__ X, _Float16* __restrict__ Y, long n) {
  long i = (long)blockIdx.x * blockDim.x + threadIdx.x;
  if (i < n) Y[i] = (_Float16)X[i];
}

// W: f32 [K][N]  ->  Wt: f16 [N][K]
__global__ void conv_wT(const float* __restrict__ W, _Float16* __restrict__ Wt,
                        int K, int N) {
  long i = (long)blockIdx.x * blockDim.x + threadIdx.x;
  if (i >= (long)N * K) return;
  int  k = (int)(i % K);
  long n = i / K;
  Wt[i] = (_Float16)W[(long)k * N + n];
}

// in: f32 [B,S,NH,D]  -> rope -> out: f16 [B,NH,S,D]
__global__ void rope_pack(const float* __restrict__ in, const float* __restrict__ cache,
                          _Float16* __restrict__ out, int NH) {
  long i = (long)blockIdx.x * blockDim.x + threadIdx.x;  // over B*S*NH*(D/2)
  long total = (long)Bn * Sn * NH * (Dn / 2);
  if (i >= total) return;
  int d2 = (int)(i & 63);           // D/2 = 64
  long t = i >> 6;
  int h  = (int)(t % NH); t /= NH;
  int s  = (int)(t % Sn);
  int b  = (int)(t / Sn);
  long ib = (((long)(b * Sn + s)) * NH + h) * Dn + 2 * d2;
  float x0 = in[ib], x1 = in[ib + 1];
  float cs = cache[(s * 64 + d2) * 2 + 0];
  float sn = cache[(s * 64 + d2) * 2 + 1];
  float re = x0 * cs - x1 * sn;
  float im = x1 * cs + x0 * sn;
  union { _Float16 h2[2]; unsigned u; } u;
  u.h2[0] = (_Float16)re; u.h2[1] = (_Float16)im;
  long ob = (((long)(b * NH + h)) * Sn + s) * Dn + 2 * d2;
  *(unsigned*)(out + ob) = u.u;   // ob even -> 4B aligned
}

// in: f32 [B,S,KVH,D] -> out: f16 [B,KVH,D,S] (transposed for P@V B-fragments)
__global__ void pack_vt(const float* __restrict__ in, _Float16* __restrict__ out) {
  long i = (long)blockIdx.x * blockDim.x + threadIdx.x;  // over B*KVH*D*S
  long total = (long)Bn * KVHn * Dn * Sn;
  if (i >= total) return;
  int s = (int)(i % Sn);
  long t = i / Sn;
  int d  = (int)(t % Dn); t /= Dn;
  int kv = (int)(t % KVHn);
  int b  = (int)(t / KVHn);
  out[i] = (_Float16)in[(((long)(b * Sn + s)) * KVHn + kv) * Dn + d];
}

// ---------------------------------------------------------------------------
// WMMA GEMM: C[M,N] f32 = A[M,K] f16 @ Bt[N,K] f16 (+bias[N])
// Block: 256 thr (8 waves). Block tile 64x256, wave tile 32x64, K step 32.
// Double-buffered LDS staged via CDNA5 async LDS-DMA (ASYNCcnt).
// ---------------------------------------------------------------------------
__global__ __launch_bounds__(256) void gemm_f16f32(
    const _Float16* __restrict__ A, const _Float16* __restrict__ Bt,
    const float* __restrict__ bias, float* __restrict__ C,
    int M, int N, int K) {
  __shared__ __align__(16) _Float16 sA[2][64 * 32];    //  2 x 4 KiB
  __shared__ __align__(16) _Float16 sB[2][256 * 32];   //  2 x 16 KiB
  const int tid  = threadIdx.x;
  const int lane = tid & 31;
  const int wave = tid >> 5;
  const int m0 = blockIdx.x * 64;
  const int n0 = blockIdx.y * 256;
  const int wm = (wave & 1) * 32;   // 2 waves along M
  const int wn = (wave >> 1) * 64;  // 4 waves along N

  v8f acc[2][4];
#pragma unroll
  for (int am = 0; am < 2; ++am)
#pragma unroll
    for (int bn = 0; bn < 4; ++bn) acc[am][bn] = (v8f){};

  // stage tiles for one K-slab into buffer `bf` (5 async ops per wave)
  auto stage = [&](int bf, int k0) {
    {
      int row = tid >> 2, ch = (tid & 3) * 8;        // A: 64x32 halves
      async_g2l_b128(&sA[bf][row * 32 + ch],
                     A + (long)(m0 + row) * K + k0 + ch);
    }
#pragma unroll
    for (int rep = 0; rep < 4; ++rep) {              // B: 256x32 halves
      int idx = tid + rep * 256;
      int row = idx >> 2, ch = (idx & 3) * 8;
      async_g2l_b128(&sB[bf][row * 32 + ch],
                     Bt + (long)(n0 + row) * K + k0 + ch);
    }
  };

  auto compute = [&](int bf) {
    v16h a0 = load_fragA(&sA[bf][(wm + 0)  * 32], 32, lane);
    v16h a1 = load_fragA(&sA[bf][(wm + 16) * 32], 32, lane);
    v16h bfr[4];
#pragma unroll
    for (int bn = 0; bn < 4; ++bn)
      bfr[bn] = load_fragB(&sB[bf][(wn + bn * 16) * 32], 32, lane);
#pragma unroll
    for (int bn = 0; bn < 4; ++bn) acc[0][bn] = wmma16(a0, bfr[bn], acc[0][bn]);
#pragma unroll
    for (int bn = 0; bn < 4; ++bn) acc[1][bn] = wmma16(a1, bfr[bn], acc[1][bn]);
  };

  stage(0, 0);
  int buf = 0;
  for (int k0 = 32; k0 < K; k0 += 32) {
    __syncthreads();                                   // buf^1 free to overwrite
    stage(buf ^ 1, k0);                                // DMA next slab
    asm volatile("s_wait_asynccnt 0x5" ::: "memory");  // previous slab landed
    __syncthreads();                                   // ... in all waves
    compute(buf);                                      // overlaps with DMA
    buf ^= 1;
  }
  __syncthreads();
  asm volatile("s_wait_asynccnt 0x0" ::: "memory");
  __syncthreads();
  compute(buf);

  // epilogue: C element (M = j + 8*(lane>>4), N = lane&15) per 16x16 tile
  int nl  = lane & 15;
  int mh8 = (lane >> 4) * 8;
#pragma unroll
  for (int am = 0; am < 2; ++am)
#pragma unroll
    for (int bn = 0; bn < 4; ++bn) {
      int ng = n0 + wn + bn * 16 + nl;
      float bb = bias ? bias[ng] : 0.0f;
#pragma unroll
      for (int j = 0; j < 8; ++j) {
        int mg = m0 + wm + am * 16 + mh8 + j;
        C[(long)mg * N + ng] = acc[am][bn][j] + bb;
      }
    }
}

// ---------------------------------------------------------------------------
// Flash attention: one wave per 16-row Q tile, key blocks of 32, causal.
// Q: f16 [B,H,S,D]  K: f16 [B,KVH,S,D]  Vt: f16 [B,KVH,D,S]
// O: f16 [B,S,H*D]  (feeds final GEMM)
// ---------------------------------------------------------------------------
__global__ __launch_bounds__(256) void flash_attn(
    const _Float16* __restrict__ Q, const _Float16* __restrict__ Kt,
    const _Float16* __restrict__ Vt, _Float16* __restrict__ O) {
  __shared__ __align__(16) _Float16 sP[8][16 * 32];  // per-wave P staging
  const int lane = threadIdx.x & 31;
  const int wave = threadIdx.x >> 5;
  const int t  = blockIdx.x * 8 + wave;          // q-tile id, B*H*(S/16) total
  const int qt = t & (Sn / 16 - 1);              // & 127
  const int h  = (t >> 7) & (Hn - 1);
  const int b  = t >> 11;
  const int q0 = qt * 16;
  const int kvh = h / (Hn / KVHn);

  const _Float16* qbase = Q  + (((long)(b * Hn   + h  )) * Sn + q0) * Dn;
  const _Float16* kbase = Kt + (((long)(b * KVHn + kvh)) * Sn) * Dn;
  const _Float16* vbase = Vt + (((long)(b * KVHn + kvh)) * Dn) * Sn;

  // Q fragments: 4 slabs of K=32 over D=128, kept resident
  v16h qf[4];
#pragma unroll
  for (int sl = 0; sl < 4; ++sl) qf[sl] = load_fragA(qbase + sl * 32, Dn, lane);

  v8f o[8];
#pragma unroll
  for (int f = 0; f < 8; ++f) o[f] = (v8f){};
  float mrow[8], lrow[8];
#pragma unroll
  for (int j = 0; j < 8; ++j) { mrow[j] = -1e30f; lrow[j] = 0.0f; }

  _Float16* myP = &sP[wave][0];
  const float scale = 0.08838834764831845f;  // 1/sqrt(128)
  const int mh8 = (lane >> 4) * 8;
  const int nl  = lane & 15;

  const int nkb = (q0 + 16 + 31) >> 5;  // key blocks covering keys 0..q0+15
  for (int kb = 0; kb < nkb; ++kb) {
    const int k0 = kb * 32;
    // ---- scores: S(16x32) = Q(16x128) @ K^T, two 16-col halves ----
    v8f s0 = {}, s1 = {};
#pragma unroll
    for (int sl = 0; sl < 4; ++sl) {
      v16h bk0 = load_fragB(kbase + (long)(k0 +  0) * Dn + sl * 32, Dn, lane);
      v16h bk1 = load_fragB(kbase + (long)(k0 + 16) * Dn + sl * 32, Dn, lane);
      s0 = wmma16(qf[sl], bk0, s0);
      s1 = wmma16(qf[sl], bk1, s1);
    }
    // ---- scale + causal mask + online softmax ----
    float al[8];
#pragma unroll
    for (int j = 0; j < 8; ++j) {
      int qg = q0 + mh8 + j;
      int n0g = k0 + nl;
      float x0 = s0[j] * scale; if (n0g      > qg) x0 = -1e30f;
      float x1 = s1[j] * scale; if (n0g + 16 > qg) x1 = -1e30f;
      float mx = fmaxf(x0, x1);
      mx = fmaxf(mx, __shfl_xor(mx, 1, 32));
      mx = fmaxf(mx, __shfl_xor(mx, 2, 32));
      mx = fmaxf(mx, __shfl_xor(mx, 4, 32));
      mx = fmaxf(mx, __shfl_xor(mx, 8, 32));
      float mn = fmaxf(mrow[j], mx);
      float a  = __expf(mrow[j] - mn);
      float p0 = __expf(x0 - mn);
      float p1 = __expf(x1 - mn);
      float rs = p0 + p1;
      rs += __shfl_xor(rs, 1, 32);
      rs += __shfl_xor(rs, 2, 32);
      rs += __shfl_xor(rs, 4, 32);
      rs += __shfl_xor(rs, 8, 32);
      lrow[j] = lrow[j] * a + rs;
      mrow[j] = mn;
      al[j] = a;
      s0[j] = p0; s1[j] = p1;
    }
#pragma unroll
    for (int f = 0; f < 8; ++f)
#pragma unroll
      for (int j = 0; j < 8; ++j) o[f][j] *= al[j];

    // ---- C-layout -> A-layout via per-wave LDS tile ----
#pragma unroll
    for (int j = 0; j < 8; ++j) {
      int r = mh8 + j;
      myP[r * 32 + nl]      = (_Float16)s0[j];
      myP[r * 32 + nl + 16] = (_Float16)s1[j];
    }
    asm volatile("s_wait_dscnt 0" ::: "memory");  // same-wave LDS RAW
    v16h pf = load_fragA(myP, 32, lane);

    // ---- O += P(16x32) @ V(32x128) ----
#pragma unroll
    for (int f = 0; f < 8; ++f) {
      v16h vf = load_fragB(vbase + (long)(f * 16) * Sn + k0, Sn, lane);
      o[f] = wmma16(pf, vf, o[f]);
    }
  }

  // ---- normalize + store f16 [B,S,H*D] ----
  float inv[8];
#pragma unroll
  for (int j = 0; j < 8; ++j) inv[j] = 1.0f / lrow[j];
#pragma unroll
  for (int f = 0; f < 8; ++f)
#pragma unroll
    for (int j = 0; j < 8; ++j) {
      long row = (long)b * Sn + q0 + mh8 + j;
      O[row * HD + h * Dn + f * 16 + nl] = (_Float16)(o[f][j] * inv[j]);
    }
}

// ---------------------------------------------------------------------------
// Workspace layout (bytes). Total ~116 MiB.
// ---------------------------------------------------------------------------
static constexpr size_t OFF_X16  = 0;          // 16 MiB  x f16 [M,HID]
static constexpr size_t OFF_Q16  = 16777216;   // 16 MiB  q f16 [B,H,S,D]
static constexpr size_t OFF_K16  = 33554432;   //  4 MiB  k f16 [B,KVH,S,D]
static constexpr size_t OFF_VT16 = 37748736;   //  4 MiB  v f16 [B,KVH,D,S]
static constexpr size_t OFF_AT16 = 41943040;   // 16 MiB  attn f16 [M,HD]
static constexpr size_t OFF_WT   = 58720256;   //  8 MiB  WqT then WoT f16
static constexpr size_t OFF_WKT  = 67108864;   //  2 MiB  WkT f16
static constexpr size_t OFF_WVT  = 69206016;   //  2 MiB  WvT f16
static constexpr size_t OFF_QF   = 71303168;   // 32 MiB  q f32 [M,HD]
static constexpr size_t OFF_KF   = 104857600;  //  8 MiB  k f32 [M,KVD]
static constexpr size_t OFF_VF   = 113246208;  //  8 MiB  v f32 [M,KVD]

extern "C" void kernel_launch(void* const* d_in, const int* in_sizes, int n_in,
                              void* d_out, int out_size, void* d_ws, size_t ws_size,
                              hipStream_t stream) {
  const float* x    = (const float*)d_in[0];
  // d_in[1] attention_mask: causal mask applied analytically, unused
  const float* rope = (const float*)d_in[2];
  const float* Wq   = (const float*)d_in[3];
  const float* bq   = (const float*)d_in[4];
  const float* Wk   = (const float*)d_in[5];
  const float* bk   = (const float*)d_in[6];
  const float* Wv   = (const float*)d_in[7];
  const float* bv   = (const float*)d_in[8];
  const float* Wo   = (const float*)d_in[9];
  float* out = (float*)d_out;
  char*  ws  = (char*)d_ws;

  _Float16* x16  = (_Float16*)(ws + OFF_X16);
  _Float16* q16  = (_Float16*)(ws + OFF_Q16);
  _Float16* k16  = (_Float16*)(ws + OFF_K16);
  _Float16* vt16 = (_Float16*)(ws + OFF_VT16);
  _Float16* at16 = (_Float16*)(ws + OFF_AT16);
  _Float16* WT   = (_Float16*)(ws + OFF_WT);
  _Float16* WkT  = (_Float16*)(ws + OFF_WKT);
  _Float16* WvT  = (_Float16*)(ws + OFF_WVT);
  float*    qf   = (float*)(ws + OFF_QF);
  float*    kf   = (float*)(ws + OFF_KF);
  float*    vf   = (float*)(ws + OFF_VF);

  const int T = 256;
  // 1) precision conversion / weight transposes
  conv_h <<<(int)(((long)Mrows * HIDn + T - 1) / T), T, 0, stream>>>(x, x16, (long)Mrows * HIDn);
  conv_wT<<<(int)(((long)HD  * HIDn + T - 1) / T), T, 0, stream>>>(Wq, WT,  HIDn, HD);
  conv_wT<<<(int)(((long)KVD * HIDn + T - 1) / T), T, 0, stream>>>(Wk, WkT, HIDn, KVD);
  conv_wT<<<(int)(((long)KVD * HIDn + T - 1) / T), T, 0, stream>>>(Wv, WvT, HIDn, KVD);

  // 2) QKV projections (bias fused)
  gemm_f16f32<<<dim3(Mrows / 64, HD  / 256), T, 0, stream>>>(x16, WT,  bq, qf, Mrows, HD,  HIDn);
  gemm_f16f32<<<dim3(Mrows / 64, KVD / 256), T, 0, stream>>>(x16, WkT, bk, kf, Mrows, KVD, HIDn);
  gemm_f16f32<<<dim3(Mrows / 64, KVD / 256), T, 0, stream>>>(x16, WvT, bv, vf, Mrows, KVD, HIDn);

  // 3) RoPE + layout packing
  rope_pack<<<(int)(((long)Bn * Sn * Hn   * 64 + T - 1) / T), T, 0, stream>>>(qf, rope, q16, Hn);
  rope_pack<<<(int)(((long)Bn * Sn * KVHn * 64 + T - 1) / T), T, 0, stream>>>(kf, rope, k16, KVHn);
  pack_vt  <<<(int)(((long)Bn * KVHn * Dn * Sn + T - 1) / T), T, 0, stream>>>(vf, vt16);

  // 4) causal flash attention (B*H*(S/16) = 4096 wave-tiles, 8 waves/block)
  flash_attn<<<512, 256, 0, stream>>>(q16, k16, vt16, at16);

  // 5) output projection (WoT reuses WqT slot), fp32 straight to d_out
  conv_wT<<<(int)(((long)HIDn * HD + T - 1) / T), T, 0, stream>>>(Wo, WT, HD, HIDn);
  gemm_f16f32<<<dim3(Mrows / 64, HIDn / 256), T, 0, stream>>>(at16, WT, nullptr, out, Mrows, HIDn, HD);
}
